// Decoder_51917564674681
// MI455X (gfx1250) — matrix operand order repaired and checked by reading, not compiled, not run on hardware
//
#include <hip/hip_runtime.h>
#include <hip/hip_bf16.h>

typedef __bf16 bf16_t;
typedef __attribute__((ext_vector_type(16))) __bf16 v16bf;
typedef __attribute__((ext_vector_type(8)))  __bf16 v8bf;
typedef __attribute__((ext_vector_type(8)))  float  v8f;

#define SOS_IDX 1
#define NEGINF (-1000000000.0f)

// sched_group_barrier masks
#define SGB_WMMA     0x0008
#define SGB_VMEM_RD  0x0020

// ---------------------------------------------------------------------------
// helpers
// ---------------------------------------------------------------------------
static __device__ __forceinline__ v16bf pack16(v8bf lo, v8bf hi) {
  return __builtin_shufflevector(lo, hi, 0,1,2,3,4,5,6,7,8,9,10,11,12,13,14,15);
}

static __device__ __forceinline__ v8f wmma_bf16(v16bf a, v16bf b, v8f c) {
  return __builtin_amdgcn_wmma_f32_16x16x32_bf16(false, a, false, b, (short)0, c,
                                                 false, false);
}

static __device__ __forceinline__ v16bf load_a16(const bf16_t* __restrict__ p) {
  return pack16(*(const v8bf*)(p), *(const v8bf*)(p + 16));
}
static __device__ __forceinline__ v16bf load_b16(const bf16_t* __restrict__ p) {
  return pack16(*(const v8bf*)(p), *(const v8bf*)(p + 16));
}

// One GEMM region, fully unrolled + software-pipelined: all fragments of
// chunk k+1 are loaded before the 4 WMMAs of chunk k. sched_group_barrier
// pins this order pre-RA ([10 VMEM-read][4 WMMA] per chunk), so the wait
// before each WMMA group leaves a whole chunk's loads in flight instead of
// draining loadcnt to 0.
// Pointers are pre-offset:
//   ap  : A base + row*astride + kg*8      (two 16B loads / chunk, +0 / +16)
//   bpX : W base + wrow*bstride + kg*8     (two 16B loads / chunk, +0 / +16)
template <int NC>
static __device__ __forceinline__ void gemm_region(
    const bf16_t* __restrict__ ap,
    const bf16_t* __restrict__ bp0, const bf16_t* __restrict__ bp1,
    const bf16_t* __restrict__ bp2, const bf16_t* __restrict__ bp3,
    v8f& a0, v8f& a1, v8f& a2, v8f& a3) {
  v16bf a  = load_a16(ap);
  v16bf b0 = load_b16(bp0);
  v16bf b1 = load_b16(bp1);
  v16bf b2 = load_b16(bp2);
  v16bf b3 = load_b16(bp3);
#pragma unroll
  for (int kc = 0; kc < NC; ++kc) {
    v16bf an = a, b0n = b0, b1n = b1, b2n = b2, b3n = b3;
    if (kc + 1 < NC) {                      // compile-time after unroll
      const int k = (kc + 1) * 32;
      an  = load_a16(ap + k);
      b0n = load_b16(bp0 + k);
      b1n = load_b16(bp1 + k);
      b2n = load_b16(bp2 + k);
      b3n = load_b16(bp3 + k);
    }
    a0 = wmma_bf16(a, b0, a0);
    a1 = wmma_bf16(a, b1, a1);
    a2 = wmma_bf16(a, b2, a2);
    a3 = wmma_bf16(a, b3, a3);
    a = an; b0 = b0n; b1 = b1n; b2 = b2n; b3 = b3n;
    // schedule spec for this chunk: next chunk's 10 loads, then 4 WMMAs
    __builtin_amdgcn_sched_group_barrier(SGB_VMEM_RD, 10, 0);
    __builtin_amdgcn_sched_group_barrier(SGB_WMMA,     4, 0);
  }
}

static __device__ __forceinline__ float sigmoidf_(float x) {
  return 1.0f / (1.0f + __expf(-x));
}
static __device__ __forceinline__ float tanhf_(float x) {
  x = fminf(15.0f, fmaxf(-15.0f, x));
  float e = __expf(2.0f * x);
  return (e - 1.0f) / (e + 1.0f);
}

// ---------------------------------------------------------------------------
// prologue kernels
// ---------------------------------------------------------------------------
__global__ void f32_to_bf16_kernel(const float* __restrict__ src,
                                   bf16_t* __restrict__ dst, int count) {
  int i = blockIdx.x * blockDim.x + threadIdx.x;
  if (i < count) dst[i] = (bf16_t)src[i];
}

// xs[t][n][e]: t==0 -> emb[SOS], else emb[text[n][t-1]]   (128 x 32 x 256)
__global__ void embed_kernel(const int* __restrict__ text,
                             const float* __restrict__ emb,
                             bf16_t* __restrict__ xs_bf) {
  int idx = blockIdx.x * blockDim.x + threadIdx.x;   // < 128*32*256
  int t  = idx >> 13;            // /(32*256)
  int r  = idx & 8191;
  int nn = r >> 8;
  int e  = r & 255;
  int tok = (t == 0) ? SOS_IDX : text[nn * 128 + (t - 1)];
  xs_bf[idx] = (bf16_t)emb[(size_t)tok * 256 + e];
}

__global__ void init_state_kernel(const float* __restrict__ values,
                                  float* __restrict__ c1, float* __restrict__ c2,
                                  bf16_t* __restrict__ h1b0, bf16_t* __restrict__ h2b0,
                                  bf16_t* __restrict__ ctxb) {
  int i = blockIdx.x * blockDim.x + threadIdx.x;     // < 32*1024
  if (i < 32 * 1024) { c1[i] = 0.0f; h1b0[i] = (bf16_t)0.0f; }
  if (i < 32 * 128) {
    c2[i] = 0.0f;
    h2b0[i] = (bf16_t)0.0f;
    int nn = i >> 7, j = i & 127;
    ctxb[i] = (bf16_t)values[(size_t)nn * 512 * 128 + j];   // values[:,0,:]
  }
}

// ---------------------------------------------------------------------------
// LSTM1: gates = [x|ctx|h1] @ [W_ih1|W_hh1].T   (32 x 4096, K = 1408)
// 128 waves; each wave: one 16x16 M-tile x 4 gate tiles (i,f,g,o) -> fused cell
// ---------------------------------------------------------------------------
__global__ void lstm1_kernel(const bf16_t* __restrict__ xs_t,   // 32x256
                             const bf16_t* __restrict__ ctxb,   // 32x128
                             const bf16_t* __restrict__ h1_old, // 32x1024
                             const bf16_t* __restrict__ Wih,    // 4096x384
                             const bf16_t* __restrict__ Whh,    // 4096x1024
                             const float* __restrict__ bih, const float* __restrict__ bhh,
                             float* __restrict__ c1,            // 32x1024
                             bf16_t* __restrict__ h1_new) {     // 32x1024
  const int lane = threadIdx.x & 31;
  const int w    = (blockIdx.x * blockDim.x + threadIdx.x) >> 5;  // 0..127
  const int m0   = (w & 1) * 16;
  const int j0   = (w >> 1) * 16;       // 0..1008
  const int n    = lane & 15;
  const int kg   = lane >> 4;
  const int arow = m0 + n;
  const int a8   = kg * 8;
  const int b16  = kg * 8;
  const int r0   = j0 + n;

  // pre-offset base pointers (computed once; all K-advances are immediates)
  const bf16_t* ax = xs_t   + (size_t)arow * 256  + a8;
  const bf16_t* ac = ctxb   + (size_t)arow * 128  + a8;
  const bf16_t* ah = h1_old + (size_t)arow * 1024 + a8;
  const bf16_t* bx0 = Wih + (size_t)(0 * 1024 + r0) * 384 + b16;
  const bf16_t* bx1 = Wih + (size_t)(1 * 1024 + r0) * 384 + b16;
  const bf16_t* bx2 = Wih + (size_t)(2 * 1024 + r0) * 384 + b16;
  const bf16_t* bx3 = Wih + (size_t)(3 * 1024 + r0) * 384 + b16;
  const bf16_t* bh0 = Whh + (size_t)(0 * 1024 + r0) * 1024 + b16;
  const bf16_t* bh1 = Whh + (size_t)(1 * 1024 + r0) * 1024 + b16;
  const bf16_t* bh2 = Whh + (size_t)(2 * 1024 + r0) * 1024 + b16;
  const bf16_t* bh3 = Whh + (size_t)(3 * 1024 + r0) * 1024 + b16;

  v8f acc0 = {}, acc1 = {}, acc2 = {}, acc3 = {};
  gemm_region<8>(ax, bx0, bx1, bx2, bx3, acc0, acc1, acc2, acc3);        // x   K 0..255
  gemm_region<4>(ac, bx0 + 256, bx1 + 256, bx2 + 256, bx3 + 256,
                 acc0, acc1, acc2, acc3);                                 // ctx K 256..383
  gemm_region<32>(ah, bh0, bh1, bh2, bh3, acc0, acc1, acc2, acc3);       // h1  K 384..1407

  const int j = j0 + n;
  const float bi  = bih[j]        + bhh[j];
  const float bf_ = bih[1024 + j] + bhh[1024 + j];
  const float bg  = bih[2048 + j] + bhh[2048 + j];
  const float bo  = bih[3072 + j] + bhh[3072 + j];
#pragma unroll
  for (int v = 0; v < 8; ++v) {
    const int row = m0 + v + 8 * kg;                // D layout: M = v + 8*(lane>>4)
    const size_t idx = (size_t)row * 1024 + j;
    const float iv = sigmoidf_(acc0[v] + bi);
    const float fv = sigmoidf_(acc1[v] + bf_);
    const float gv = tanhf_   (acc2[v] + bg);
    const float ov = sigmoidf_(acc3[v] + bo);
    const float cn = fv * c1[idx] + iv * gv;
    c1[idx] = cn;
    h1_new[idx] = (bf16_t)(ov * tanhf_(cn));
  }
}

// ---------------------------------------------------------------------------
// LSTM2: gates = [h1|h2] @ [W_ih2|W_hh2].T   (32 x 512, K = 1152), 16 waves
// ---------------------------------------------------------------------------
__global__ void lstm2_kernel(const bf16_t* __restrict__ h1_new, // 32x1024
                             const bf16_t* __restrict__ h2_old, // 32x128
                             const bf16_t* __restrict__ Wih,    // 512x1024
                             const bf16_t* __restrict__ Whh,    // 512x128
                             const float* __restrict__ bih, const float* __restrict__ bhh,
                             float* __restrict__ c2,            // 32x128
                             bf16_t* __restrict__ h2_new,       // 32x128
                             float* __restrict__ h2f) {         // 32x128
  const int lane = threadIdx.x & 31;
  const int w    = (blockIdx.x * blockDim.x + threadIdx.x) >> 5;  // 0..15
  const int m0   = (w & 1) * 16;
  const int j0   = (w >> 1) * 16;       // 0..112
  const int n    = lane & 15;
  const int kg   = lane >> 4;
  const int arow = m0 + n;
  const int a8   = kg * 8;
  const int b16  = kg * 8;
  const int r0   = j0 + n;

  const bf16_t* a1p = h1_new + (size_t)arow * 1024 + a8;
  const bf16_t* a2p = h2_old + (size_t)arow * 128  + a8;
  const bf16_t* bi0 = Wih + (size_t)(0 * 128 + r0) * 1024 + b16;
  const bf16_t* bi1 = Wih + (size_t)(1 * 128 + r0) * 1024 + b16;
  const bf16_t* bi2 = Wih + (size_t)(2 * 128 + r0) * 1024 + b16;
  const bf16_t* bi3 = Wih + (size_t)(3 * 128 + r0) * 1024 + b16;
  const bf16_t* bh0 = Whh + (size_t)(0 * 128 + r0) * 128 + b16;
  const bf16_t* bh1 = Whh + (size_t)(1 * 128 + r0) * 128 + b16;
  const bf16_t* bh2 = Whh + (size_t)(2 * 128 + r0) * 128 + b16;
  const bf16_t* bh3 = Whh + (size_t)(3 * 128 + r0) * 128 + b16;

  v8f acc0 = {}, acc1 = {}, acc2 = {}, acc3 = {};
  gemm_region<32>(a1p, bi0, bi1, bi2, bi3, acc0, acc1, acc2, acc3);  // h1 K 0..1023
  gemm_region<4>(a2p, bh0, bh1, bh2, bh3, acc0, acc1, acc2, acc3);   // h2 K 1024..1151

  const int j = j0 + n;
  const float bi  = bih[j]       + bhh[j];
  const float bf_ = bih[128 + j] + bhh[128 + j];
  const float bg  = bih[256 + j] + bhh[256 + j];
  const float bo  = bih[384 + j] + bhh[384 + j];
#pragma unroll
  for (int v = 0; v < 8; ++v) {
    const int row = m0 + v + 8 * kg;
    const size_t idx = (size_t)row * 128 + j;
    const float iv = sigmoidf_(acc0[v] + bi);
    const float fv = sigmoidf_(acc1[v] + bf_);
    const float gv = tanhf_   (acc2[v] + bg);
    const float ov = sigmoidf_(acc3[v] + bo);
    const float cn = fv * c2[idx] + iv * gv;
    c2[idx] = cn;
    const float h = ov * tanhf_(cn);
    h2_new[idx] = (bf16_t)h;
    h2f[idx] = h;
  }
}

// ---------------------------------------------------------------------------
// Attention: energy = key . h2, mask, softmax over T=512, ctx = attn @ values
// One workgroup (256 threads) per batch row.  All f32.
// ---------------------------------------------------------------------------
__global__ void attn_kernel(const float* __restrict__ key,     // 32x512x128
                            const float* __restrict__ values,  // 32x512x128
                            const int* __restrict__ lens,
                            const float* __restrict__ h2f,     // 32x128
                            bf16_t* __restrict__ ctxb) {       // 32x128
  __shared__ float h2s[128];
  __shared__ float att[512];
  __shared__ float red[256];
  const int nn  = blockIdx.x;
  const int tid = threadIdx.x;

  if (tid < 128) h2s[tid] = h2f[nn * 128 + tid];
  __syncthreads();

  const int len = lens[nn];
  for (int tt = tid; tt < 512; tt += 256) {
    const float* kp = key + ((size_t)nn * 512 + tt) * 128;
    float e = 0.0f;
#pragma unroll 8
    for (int j = 0; j < 128; ++j) e += kp[j] * h2s[j];
    att[tt] = (tt >= len) ? NEGINF : e;
  }
  __syncthreads();

  // max reduction over 512
  red[tid] = fmaxf(att[tid], att[tid + 256]);
  __syncthreads();
  for (int s = 128; s > 0; s >>= 1) {
    if (tid < s) red[tid] = fmaxf(red[tid], red[tid + s]);
    __syncthreads();
  }
  const float mx = red[0];
  __syncthreads();

  // exp + sum reduction
  const float p0 = __expf(att[tid] - mx);
  const float p1 = __expf(att[tid + 256] - mx);
  att[tid] = p0;
  att[tid + 256] = p1;
  red[tid] = p0 + p1;
  __syncthreads();
  for (int s = 128; s > 0; s >>= 1) {
    if (tid < s) red[tid] += red[tid + s];
    __syncthreads();
  }
  const float inv = 1.0f / red[0];

  if (tid < 128) {
    float c = 0.0f;
    const float* vp = values + (size_t)nn * 512 * 128 + tid;
    for (int tt = 0; tt < 512; ++tt) c += att[tt] * vp[(size_t)tt * 128];
    ctxb[nn * 128 + tid] = (bf16_t)(c * inv);
  }
}

// ---------------------------------------------------------------------------
// Output projection: pred = [h2|ctx] @ W_out.T + b_out   (32 x 8000, K = 256)
// 250 waves; each wave: one M-tile x 4 N-tiles sharing the A fragment.
// ---------------------------------------------------------------------------
__global__ void outproj_kernel(const bf16_t* __restrict__ h2b,   // 32x128
                               const bf16_t* __restrict__ ctxb,  // 32x128
                               const bf16_t* __restrict__ Wout,  // 8000x256
                               const float* __restrict__ bout,
                               float* __restrict__ dout,         // (32,128,8000)
                               int t) {
  const int lane = threadIdx.x & 31;
  const int w    = (blockIdx.x * blockDim.x + threadIdx.x) >> 5;
  if (w >= 250) return;                        // wave-uniform: EXEC stays all-1
  const int m0   = (w & 1) * 16;
  const int grp  = w >> 1;                     // 0..124 -> 64 vocab cols each
  const int n    = lane & 15;
  const int kg   = lane >> 4;
  const int arow = m0 + n;
  const int a8   = kg * 8;
  const int b16  = kg * 8;

  const bf16_t* ahp = h2b  + (size_t)arow * 128 + a8;
  const bf16_t* acp = ctxb + (size_t)arow * 128 + a8;
  const bf16_t* bw0 = Wout + (size_t)(grp * 64 + 0 * 16 + n) * 256 + b16;
  const bf16_t* bw1 = Wout + (size_t)(grp * 64 + 1 * 16 + n) * 256 + b16;
  const bf16_t* bw2 = Wout + (size_t)(grp * 64 + 2 * 16 + n) * 256 + b16;
  const bf16_t* bw3 = Wout + (size_t)(grp * 64 + 3 * 16 + n) * 256 + b16;

  v8f acc0 = {}, acc1 = {}, acc2 = {}, acc3 = {};
  gemm_region<4>(ahp, bw0, bw1, bw2, bw3, acc0, acc1, acc2, acc3);       // h2  K 0..127
  gemm_region<4>(acp, bw0 + 128, bw1 + 128, bw2 + 128, bw3 + 128,
                 acc0, acc1, acc2, acc3);                                 // ctx K 128..255

  float* orow = dout + (size_t)t * 8000 + grp * 64 + n;
  v8f accs[4] = {acc0, acc1, acc2, acc3};
#pragma unroll
  for (int i = 0; i < 4; ++i) {
    const int col = grp * 64 + i * 16 + n;
    const float bb = bout[col];
#pragma unroll
    for (int v = 0; v < 8; ++v) {
      const int row = m0 + v + 8 * kg;
      orow[(size_t)row * 128 * 8000 + i * 16] = accs[i][v] + bb;
    }
  }
}

// ---------------------------------------------------------------------------
// host launcher
// ---------------------------------------------------------------------------
extern "C" void kernel_launch(void* const* d_in, const int* in_sizes, int n_in,
                              void* d_out, int out_size, void* d_ws, size_t ws_size,
                              hipStream_t stream) {
  (void)in_sizes; (void)n_in; (void)out_size; (void)ws_size;
  const float* key    = (const float*)d_in[0];
  const float* values = (const float*)d_in[1];
  const int*   lens   = (const int*)d_in[2];
  const int*   text   = (const int*)d_in[3];
  const float* emb    = (const float*)d_in[4];
  const float* Wih1   = (const float*)d_in[5];
  const float* Whh1   = (const float*)d_in[6];
  const float* bih1   = (const float*)d_in[7];
  const float* bhh1   = (const float*)d_in[8];
  const float* Wih2   = (const float*)d_in[9];
  const float* Whh2   = (const float*)d_in[10];
  const float* bih2   = (const float*)d_in[11];
  const float* bhh2   = (const float*)d_in[12];
  const float* Wout   = (const float*)d_in[13];
  const float* bout   = (const float*)d_in[14];
  float* dout = (float*)d_out;

  char* ws = (char*)d_ws;
  size_t off = 0;
  auto alloc = [&](size_t bytes) -> void* {
    void* p = (void*)(ws + off);
    off = (off + bytes + 255) & ~(size_t)255;
    return p;
  };

  bf16_t* xs_bf = (bf16_t*)alloc((size_t)128 * 32 * 256 * 2);
  bf16_t* Wih1b = (bf16_t*)alloc((size_t)4096 * 384 * 2);
  bf16_t* Whh1b = (bf16_t*)alloc((size_t)4096 * 1024 * 2);
  bf16_t* Wih2b = (bf16_t*)alloc((size_t)512 * 1024 * 2);
  bf16_t* Whh2b = (bf16_t*)alloc((size_t)512 * 128 * 2);
  bf16_t* Woutb = (bf16_t*)alloc((size_t)8000 * 256 * 2);
  bf16_t* h1b   = (bf16_t*)alloc((size_t)2 * 32 * 1024 * 2);  // double-buffered
  bf16_t* h2b   = (bf16_t*)alloc((size_t)2 * 32 * 128 * 2);   // double-buffered
  bf16_t* ctxb  = (bf16_t*)alloc((size_t)32 * 128 * 2);
  float*  c1    = (float*)alloc((size_t)32 * 1024 * 4);
  float*  c2    = (float*)alloc((size_t)32 * 128 * 4);
  float*  h2f   = (float*)alloc((size_t)32 * 128 * 4);

  // ---- prologue (runs every call; deterministic) ----
  auto cvt = [&](const float* s, bf16_t* d, int cnt) {
    f32_to_bf16_kernel<<<dim3((cnt + 255) / 256), dim3(256), 0, stream>>>(s, d, cnt);
  };
  cvt(Wih1, Wih1b, 4096 * 384);
  cvt(Whh1, Whh1b, 4096 * 1024);
  cvt(Wih2, Wih2b, 512 * 1024);
  cvt(Whh2, Whh2b, 512 * 128);
  cvt(Wout, Woutb, 8000 * 256);
  embed_kernel<<<dim3(4096), dim3(256), 0, stream>>>(text, emb, xs_bf);
  init_state_kernel<<<dim3(128), dim3(256), 0, stream>>>(values, c1, c2,
                                                         h1b, h2b, ctxb);

  // ---- 128 sequential decode steps ----
  for (int t = 0; t < 128; ++t) {
    const int cur = t & 1;
    const int nxt = cur ^ 1;
    bf16_t* h1_old = h1b + (size_t)cur * 32 * 1024;
    bf16_t* h1_new = h1b + (size_t)nxt * 32 * 1024;
    bf16_t* h2_old = h2b + (size_t)cur * 32 * 128;
    bf16_t* h2_new = h2b + (size_t)nxt * 32 * 128;

    lstm1_kernel<<<dim3(16), dim3(256), 0, stream>>>(
        xs_bf + (size_t)t * 32 * 256, ctxb, h1_old, Wih1b, Whh1b, bih1, bhh1,
        c1, h1_new);
    lstm2_kernel<<<dim3(2), dim3(256), 0, stream>>>(
        h1_new, h2_old, Wih2b, Whh2b, bih2, bhh2, c2, h2_new, h2f);
    attn_kernel<<<dim3(32), dim3(256), 0, stream>>>(key, values, lens, h2f, ctxb);
    outproj_kernel<<<dim3(32), dim3(256), 0, stream>>>(h2_new, ctxb, Woutb, bout,
                                                       dout, t);
  }
}
